// UBlock_82179904242419
// MI455X (gfx1250) — compile-verified, matrix operand order checked by reading, not run on hardware
//
#include <hip/hip_runtime.h>
#include <functional>
#include <algorithm>

// ---------------------------------------------------------------------------
// CDNA5 (gfx1250) sparse-conv U-Net.  Core compute: gather-GEMM with
// v_wmma_f32_16x16x32_bf16, wave32, 64x16 macro-tile per wave:
//   - 4 independent accumulators (4-deep independent WMMA issue)
//   - loads grouped two WMMA-steps ahead (20 loads, then 8 WMMAs); a
//     sched_barrier pins that shape -> staggered s_wait_loadcnt 0xe..0x0
//   - invalid taps redirected to a zeroed pad via one 32-bit offset cndmask
//   - GATHER template param: dense (Wi) instantiations carry no nbr checks,
//     gather instantiations dereference nbr branch-free
//   - 32-bit byte offsets from uniform bases (saddr loads), strength-reduced
//     weight offsets -> zero address VALU inside a load/WMMA group
// Inverse conv = WMMA + f32 atomic scatter.  Working set is L2-resident.
// ---------------------------------------------------------------------------

typedef __bf16 bf16;
typedef __attribute__((ext_vector_type(16))) __bf16   v16bf;
typedef __attribute__((ext_vector_type(8)))  float    v8f;
typedef __attribute__((ext_vector_type(4)))  unsigned u32x4;

#if __has_builtin(__builtin_amdgcn_sched_barrier)
#define SCHED_FENCE() __builtin_amdgcn_sched_barrier(0)
#else
#define SCHED_FENCE()
#endif

#define BN_EPS 1e-4f

static inline size_t ceil_div(size_t a, size_t b) { return (a + b - 1) / b; }

// ------------------------------- utility kernels ---------------------------

__global__ void k_zero_f32(float* p, int n) {
  int i = blockIdx.x * blockDim.x + threadIdx.x;
  if (i < n) p[i] = 0.0f;
}

__global__ void k_copy_f32(const float* __restrict__ a, float* __restrict__ b, int n) {
  int i = blockIdx.x * blockDim.x + threadIdx.x;
  if (i < n) b[i] = a[i];
}

__global__ void k_cvt_bf16(const float* __restrict__ a, bf16* __restrict__ b, int n) {
  int i = blockIdx.x * blockDim.x + threadIdx.x;
  if (i < n) b[i] = (bf16)a[i];
}

// copy [N,Cs] dense block into columns [off, off+Cs) of a [N,ldd] buffer
__global__ void k_copy_cols(const float* __restrict__ s, float* __restrict__ d,
                            int N, int Cs, int ldd, int off) {
  int i = blockIdx.x * blockDim.x + threadIdx.x;
  if (i >= N * Cs) return;
  int c = i % Cs;
  int r = i / Cs;
  d[r * ldd + off + c] = s[i];
}

// ------------------------- weight pack (B-fragment order) ------------------
// v_wmma_*_16x16x32 B layout (CDNA5 ISA 7.12.2, 16-bit operand): lane L holds
// column n=L%16; its 16 K-elements are, with half=L/16:
//   K in [8*half, 8*half+8) ++ [16+8*half, 16+8*half+8)
// Pack W[k][cin][cout] so each lane reads 16 contiguous bf16 (32B, aligned).
__global__ void k_pack_w(const float* __restrict__ W, bf16* __restrict__ Wp,
                         int K, int Cin, int Cout) {
  int i = blockIdx.x * blockDim.x + threadIdx.x;
  if (i >= K * Cin * Cout) return;
  int cout = i % Cout;
  int t    = i / Cout;
  int cin  = t % Cin;
  int k    = t / Cin;
  int ch   = cin >> 5;        // 32-wide K chunk
  int kl   = cin & 31;
  int half = (kl >> 3) & 1;   // chunks {0-7,16-23}->half0, {8-15,24-31}->half1
  int base = kl - 8 * half;
  int e    = (base < 8) ? base : base - 8;
  int dst  = ((k * (Cin >> 5) + ch) * Cout + cout) * 32 + half * 16 + e;
  Wp[dst] = (bf16)W[i];
}

// ------------------------------ BatchNorm ----------------------------------

// blockDim.x == C: thread owns one channel, rows coalesced across lanes,
// register accumulation, single global f32 atomic per thread at the end.
__global__ void k_bn_stats(const float* __restrict__ x, int N, int C,
                           float* __restrict__ sums, float* __restrict__ sqs) {
  int c = threadIdx.x;
  float s = 0.0f, q = 0.0f;
  for (int r = blockIdx.x; r < N; r += gridDim.x) {
    float v = x[(size_t)r * C + c];
    s += v;
    q += v * v;
  }
  atomicAdd(&sums[c], s);
  atomicAdd(&sqs[c], q);
}

// y = bf16(relu((x-mu)*rsqrt(var+eps)*g + b)), biased var (matches jnp.var)
__global__ void k_bn_relu_bf16(const float* __restrict__ x,
                               const float* __restrict__ sums, const float* __restrict__ sqs,
                               const float* __restrict__ g, const float* __restrict__ b,
                               bf16* __restrict__ y, int N, int C) {
  int i = blockIdx.x * blockDim.x + threadIdx.x;
  if (i >= N * C) return;
  int c = i % C;
  float invN = 1.0f / (float)N;
  float mu  = sums[c] * invN;
  float var = sqs[c] * invN - mu * mu;
  float v = (x[i] - mu) * rsqrtf(var + BN_EPS) * g[c] + b[c];
  y[i] = (bf16)fmaxf(v, 0.0f);
}

// --------------------------- gather-GEMM (WMMA) ----------------------------
// One wave computes a 64x16 f32 macro-tile of out[Nout, :].  NCH = Cin/32,
// COUT = Cout (compile-time -> all chunk offsets are immediates).
// GATHER=true: rows gathered via nbr (branch-free); invalid taps (nbr<0 /
// row overhang) load from the zeroed pad at byte offset `zoff` inside X, so
// EXEC stays all-1s at every WMMA.  GATHER=false: dense rows (1x1 Wi GEMM).
// addin (optional, stride lda) fuses the residual add into the store.
template <int NCH, int COUT, bool GATHER>
__global__ __launch_bounds__(32)
void k_gather_gemm(const bf16* __restrict__ X, unsigned zoff,
                   const bf16* __restrict__ Wp, const int* __restrict__ nbr, int Ktaps,
                   float* __restrict__ out, int Nout,
                   int ldo, int ocol, const float* __restrict__ addin, int lda) {
  const int lane  = threadIdx.x;
  const int n16   = lane & 15;
  const int half  = lane >> 4;
  const int row0  = blockIdx.x * 64;
  const int cout0 = blockIdx.y * 16;
  const unsigned rowB = NCH * 64u;                  // activation row bytes
  const unsigned hb   = 16u * (unsigned)half;
  const unsigned zph  = zoff + hb;                  // pad offset incl. lane term
  const unsigned wstride = (unsigned)NCH * COUT * 64u;  // W bytes per tap
  const char* Xb = (const char*)X;
  const char* Wb = (const char*)Wp;
  v8f acc[4] = {{}, {}, {}, {}};

  auto tapaddr = [&](int k, unsigned ao[4]) {
#pragma unroll
    for (int i = 0; i < 4; ++i) {
      int r  = row0 + 16 * i + n16;
      int rr = r < Nout ? r : Nout - 1;
      int s;
      if constexpr (GATHER)
        s = nbr[(unsigned)rr * Ktaps + k];
      else
        s = rr;
      if (r >= Nout) s = -1;
      ao[i] = (s >= 0) ? ((unsigned)s * rowB + hb) : zph;   // mad + cndmask
    }
  };
  auto ldB = [&](unsigned wk, int ch, u32x4 B[2]) {
    B[0] = *(const u32x4*)(Wb + wk + ch * COUT * 64);
    B[1] = *(const u32x4*)(Wb + wk + ch * COUT * 64 + 16);
  };
  auto ldA = [&](const unsigned ao[4], int ch, u32x4 A[4][2]) {
#pragma unroll
    for (int i = 0; i < 4; ++i) {
      A[i][0] = *(const u32x4*)(Xb + ao[i] + ch * 64);
      A[i][1] = *(const u32x4*)(Xb + ao[i] + ch * 64 + 32);
    }
  };
  auto mm = [&](const u32x4 B[2], const u32x4 A[4][2]) {
    union { u32x4 u[2]; v16bf v; } bv;
    bv.u[0] = B[0]; bv.u[1] = B[1];
#pragma unroll
    for (int i = 0; i < 4; ++i) {
      union { u32x4 u[2]; v16bf v; } av;
      av.u[0] = A[i][0]; av.u[1] = A[i][1];
      acc[i] = __builtin_amdgcn_wmma_f32_16x16x32_bf16(
          false, av.v, false, bv.v, (short)0, acc[i], false, false);
    }
  };

  unsigned wk = (unsigned)(cout0 + n16) * 64u + 32u * half;   // tap-0 W voffset
  if constexpr (NCH == 1) {
    // two taps unrolled: 20 loads issued, then 8 WMMAs
    int k = 0;
    for (; k + 2 <= Ktaps; k += 2, wk += 2 * wstride) {
      unsigned a0[4], a1[4];
      tapaddr(k, a0);
      tapaddr(k + 1, a1);
      u32x4 B0[2], B1[2], A0[4][2], A1[4][2];
      ldB(wk, 0, B0);
      ldB(wk + wstride, 0, B1);
      ldA(a0, 0, A0);
      ldA(a1, 0, A1);
      SCHED_FENCE();                 // keep all 20 loads ahead of the WMMAs
      mm(B0, A0);
      mm(B1, A1);
    }
    if (k < Ktaps) {
      unsigned a0[4];
      tapaddr(k, a0);
      u32x4 B0[2], A0[4][2];
      ldB(wk, 0, B0);
      ldA(a0, 0, A0);
      SCHED_FENCE();
      mm(B0, A0);
    }
  } else {
    // chunk pairs within a tap: 20 loads issued, then 8 WMMAs
    for (int k = 0; k < Ktaps; ++k, wk += wstride) {
      unsigned a0[4];
      tapaddr(k, a0);
#pragma unroll
      for (int g = 0; g + 2 <= NCH; g += 2) {
        u32x4 B0[2], B1[2], A0[4][2], A1[4][2];
        ldB(wk, g + 0, B0);
        ldB(wk, g + 1, B1);
        ldA(a0, g + 0, A0);
        ldA(a0, g + 1, A1);
        SCHED_FENCE();
        mm(B0, A0);
        mm(B1, A1);
      }
      if constexpr ((NCH & 1) != 0) {
        u32x4 B0[2], A0[4][2];
        ldB(wk, NCH - 1, B0);
        ldA(a0, NCH - 1, A0);
        SCHED_FENCE();
        mm(B0, A0);
      }
    }
  }

  // stores (and fused residual add) with 32-bit byte offsets
  char* Ob = (char*)out;
  const char* Ab = (const char*)addin;
  const unsigned ldoB = (unsigned)ldo * 4u;
  const unsigned ldaB = (unsigned)lda * 4u;
  const unsigned ocB  = (unsigned)(ocol + cout0 + n16) * 4u;
  const unsigned acB  = (unsigned)(cout0 + n16) * 4u;
#pragma unroll
  for (int i = 0; i < 4; ++i) {
#pragma unroll
    for (int t = 0; t < 8; ++t) {            // D: VGPR t -> row 16*i + t + 8*half
      int row = row0 + 16 * i + t + 8 * half;
      if (row < Nout) {
        float v = acc[i][t];
        if (addin) v += *(const float*)(Ab + (unsigned)row * ldaB + acB);
        *(float*)(Ob + (unsigned)row * ldoB + ocB) = v;
      }
    }
  }
}

// ---------------------- inverse conv: WMMA + atomic scatter ----------------
// For tap k (blockIdx.z): 64x16 macro-tile = X[64 rows] @ W_up[k]; rows
// scattered with f32 global atomics to out[nbr[n,k]] (masked rows dropped).
// Ndown is always a multiple of 64 here (40000/16000/6400): no row guards.
template <int NCH, int COUT>
__global__ __launch_bounds__(32)
void k_scatter_gemm(const bf16* __restrict__ X, const bf16* __restrict__ Wp,
                    const int* __restrict__ nbr, int Ktaps,
                    float* __restrict__ out, int ldo, int ocol) {
  const int lane  = threadIdx.x;
  const int n16   = lane & 15;
  const int half  = lane >> 4;
  const int row0  = blockIdx.x * 64;
  const int cout0 = blockIdx.y * 16;
  const int k     = blockIdx.z;
  const unsigned rowB = NCH * 64u;
  const unsigned hb   = 16u * (unsigned)half;
  const char* Xb = (const char*)X;
  const char* Wb = (const char*)Wp;
  v8f acc[4] = {{}, {}, {}, {}};
  const unsigned wk = (unsigned)(k * NCH) * COUT * 64u
                      + (unsigned)(cout0 + n16) * 64u + 32u * half;
  unsigned ao[4];
#pragma unroll
  for (int i = 0; i < 4; ++i)
    ao[i] = (unsigned)(row0 + 16 * i + n16) * rowB + hb;

  auto ldB = [&](int ch, u32x4 B[2]) {
    B[0] = *(const u32x4*)(Wb + wk + ch * COUT * 64);
    B[1] = *(const u32x4*)(Wb + wk + ch * COUT * 64 + 16);
  };
  auto ldA = [&](int ch, u32x4 A[4][2]) {
#pragma unroll
    for (int i = 0; i < 4; ++i) {
      A[i][0] = *(const u32x4*)(Xb + ao[i] + ch * 64);
      A[i][1] = *(const u32x4*)(Xb + ao[i] + ch * 64 + 32);
    }
  };
  auto mm = [&](const u32x4 B[2], const u32x4 A[4][2]) {
    union { u32x4 u[2]; v16bf v; } bv;
    bv.u[0] = B[0]; bv.u[1] = B[1];
#pragma unroll
    for (int i = 0; i < 4; ++i) {
      union { u32x4 u[2]; v16bf v; } av;
      av.u[0] = A[i][0]; av.u[1] = A[i][1];
      acc[i] = __builtin_amdgcn_wmma_f32_16x16x32_bf16(
          false, av.v, false, bv.v, (short)0, acc[i], false, false);
    }
  };

#pragma unroll
  for (int g = 0; g + 2 <= NCH; g += 2) {
    u32x4 B0[2], B1[2], A0[4][2], A1[4][2];
    ldB(g + 0, B0);
    ldB(g + 1, B1);
    ldA(g + 0, A0);
    ldA(g + 1, A1);
    SCHED_FENCE();
    mm(B0, A0);
    mm(B1, A1);
  }
  if constexpr ((NCH & 1) != 0) {
    u32x4 B0[2], A0[4][2];
    ldB(NCH - 1, B0);
    ldA(NCH - 1, A0);
    SCHED_FENCE();
    mm(B0, A0);
  }

#pragma unroll
  for (int i = 0; i < 4; ++i) {
#pragma unroll
    for (int t = 0; t < 8; ++t) {
      int drow = nbr[(unsigned)(row0 + 16 * i + t + 8 * half) * Ktaps + k];
      if (drow >= 0)
        atomicAdd(&out[(size_t)drow * ldo + ocol + cout0 + n16], acc[i][t]);
    }
  }
}

// ---------------------------------------------------------------------------
//                               host orchestration
// ---------------------------------------------------------------------------

extern "C" void kernel_launch(void* const* d_in, const int* in_sizes, int n_in,
                              void* d_out, int out_size, void* d_ws, size_t ws_size,
                              hipStream_t stream) {
  (void)in_sizes; (void)n_in; (void)out_size; (void)ws_size;
  static const int CL[4] = {32, 64, 96, 128};
  static const int NL[4] = {100000, 40000, 16000, 6400};

  // -------- workspace bump allocator (deterministic per call) --------
  char* ws = (char*)d_ws;
  size_t off = 0;
  auto alloc = [&](size_t bytes) -> void* {
    off = (off + 255) & ~(size_t)255;
    void* p = ws + off;
    off += bytes;
    return p;
  };

  auto Z = [&](float* p, int n) {
    k_zero_f32<<<dim3((unsigned)ceil_div(n, 256)), 256, 0, stream>>>(p, n);
  };

  // -------- pack all conv weights into bf16 B-fragment order --------
  auto F = [&](int i) -> const float* { return (const float*)d_in[i]; };
  auto packW = [&](int idx, int K, int cin, int cout) -> const bf16* {
    int tot = K * cin * cout;
    bf16* p = (bf16*)alloc((size_t)tot * sizeof(bf16));
    k_pack_w<<<dim3((unsigned)ceil_div(tot, 256)), 256, 0, stream>>>(F(idx), p, K, cin, cout);
    return p;
  };

  struct RB { const float *g1, *b1, *g2, *b2; const bf16 *W1, *W2, *Wi; int cin, cout; };
  struct Lvl {
    RB blocks[2], tails[2];
    const float *conv_g, *conv_b, *deconv_g, *deconv_b;
    const bf16 *Wdown, *Wup;
  } lvl[4];

  // Param leaves follow jax tree-flatten order (dict keys sorted):
  // per level (<3): W_down, W_up, blocks[{W1,W2,b1,b2,g1,g2}x2], conv_b, conv_g,
  //                 deconv_b, deconv_g, tail0{W1,W2,Wi,b1,b2,g1,g2}, tail1{...}
  int p = 1;  // d_in[0] = feats
  for (int l = 0; l < 4; ++l) {
    int C = CL[l];
    if (l < 3) {
      int C2 = CL[l + 1];
      lvl[l].Wdown = packW(p + 0, 8, C, C2);
      lvl[l].Wup   = packW(p + 1, 8, C2, C);
      int q = p + 2;
      for (int i = 0; i < 2; ++i, q += 6)
        lvl[l].blocks[i] = RB{F(q + 4), F(q + 2), F(q + 5), F(q + 3),
                              packW(q + 0, 27, C, C), packW(q + 1, 27, C, C),
                              nullptr, C, C};
      lvl[l].conv_b = F(q + 0); lvl[l].conv_g = F(q + 1);
      lvl[l].deconv_b = F(q + 2); lvl[l].deconv_g = F(q + 3);
      q += 4;
      lvl[l].tails[0] = RB{F(q + 5), F(q + 3), F(q + 6), F(q + 4),
                           packW(q + 0, 27, 2 * C, C), packW(q + 1, 27, C, C),
                           packW(q + 2, 1, 2 * C, C), 2 * C, C};
      q += 7;
      lvl[l].tails[1] = RB{F(q + 4), F(q + 2), F(q + 5), F(q + 3),
                           packW(q + 0, 27, C, C), packW(q + 1, 27, C, C),
                           nullptr, C, C};
      q += 6;
      p = q;
    } else {
      int q = p;
      for (int i = 0; i < 2; ++i, q += 6)
        lvl[l].blocks[i] = RB{F(q + 4), F(q + 2), F(q + 5), F(q + 3),
                              packW(q + 0, 27, C, C), packW(q + 1, 27, C, C),
                              nullptr, C, C};
      p = q;
    }
  }
  const int* nbr_subm[4] = {(const int*)d_in[p + 0], (const int*)d_in[p + 1],
                            (const int*)d_in[p + 2], (const int*)d_in[p + 3]};
  const int* nbr_down[3] = {(const int*)d_in[p + 4], (const int*)d_in[p + 5],
                            (const int*)d_in[p + 6]};

  // -------- activation buffers (bf16 buffers carry a 512B zeroed pad) ------
  float *xb[4], *yb[4], *hb[4], *ib[4], *st[4];
  bf16 *bA[4], *bB[4];
  unsigned zp[4];                           // byte offset of zero pad in bA/bB
  for (int l = 0; l < 4; ++l) {
    size_t n2 = (size_t)NL[l] * 2 * CL[l];
    size_t n1 = (size_t)NL[l] * CL[l];
    xb[l] = (float*)alloc(n2 * 4);
    yb[l] = (float*)alloc(n2 * 4);
    hb[l] = (float*)alloc(n1 * 4);
    ib[l] = (float*)alloc(n1 * 4);
    bA[l] = (bf16*)alloc(n2 * 2 + 512);
    bB[l] = (bf16*)alloc(n2 * 2 + 512);
    st[l] = (float*)alloc((size_t)4 * CL[l] * 4);   // sum + sumsq, up to 2C channels
    zp[l] = (unsigned)(n2 * 2);
    Z((float*)((char*)bA[l] + zp[l]), 128);         // zero the pads once per call
    Z((float*)((char*)bB[l] + zp[l]), 128);
  }

  // -------- op wrappers --------
  auto bnrelu = [&](const float* x, int N, int C, const float* g, const float* b,
                    bf16* y, float* stat) {
    Z(stat, 2 * C);
    unsigned blocks = (unsigned)std::min(N, 768);
    k_bn_stats<<<dim3(blocks), (unsigned)C, 0, stream>>>(x, N, C, stat, stat + C);
    k_bn_relu_bf16<<<dim3((unsigned)ceil_div((size_t)N * C, 256)), 256, 0, stream>>>(
        x, stat, stat + C, g, b, y, N, C);
  };

  auto gemm = [&](const bf16* X, unsigned zoff, const bf16* Wp, const int* nbr,
                  int Ktaps, float* out, int Nout, int Cin, int Cout, int ldo,
                  int ocol, const float* addin, int lda) {
    int nch = Cin >> 5;
    dim3 g((unsigned)ceil_div(Nout, 64), (unsigned)(Cout / 16));
#define GG(n, c)                                                                   \
    if (nbr && nch == (n) && Cout == (c)) {                                        \
      k_gather_gemm<(n), (c), true><<<g, 32, 0, stream>>>(                         \
          X, zoff, Wp, nbr, Ktaps, out, Nout, ldo, ocol, addin, lda);              \
      return;                                                                      \
    }
#define GD(n, c)                                                                   \
    if (!nbr && nch == (n) && Cout == (c)) {                                       \
      k_gather_gemm<(n), (c), false><<<g, 32, 0, stream>>>(                        \
          X, zoff, Wp, nullptr, Ktaps, out, Nout, ldo, ocol, addin, lda);          \
      return;                                                                      \
    }
    GG(1, 32) GG(2, 32) GG(1, 64) GG(2, 64) GG(4, 64)
    GG(2, 96) GG(3, 96) GG(6, 96) GG(3, 128) GG(4, 128)
    GD(2, 32) GD(4, 64) GD(6, 96)            // dense 1x1 Wi GEMMs
#undef GG
#undef GD
  };

  auto scatter = [&](const bf16* X, const bf16* Wp, const int* nbr, int Ktaps,
                     float* out, int Ndown, int Cin, int Cout, int ldo, int ocol) {
    int nch = Cin >> 5;
    dim3 g((unsigned)(Ndown / 64), (unsigned)(Cout / 16), (unsigned)Ktaps);
#define SG(n, c)                                                                   \
    if (nch == (n) && Cout == (c)) {                                               \
      k_scatter_gemm<(n), (c)><<<g, 32, 0, stream>>>(X, Wp, nbr, Ktaps, out, ldo,  \
                                                     ocol);                        \
      return;                                                                      \
    }
    SG(2, 32) SG(3, 64) SG(4, 96)
#undef SG
  };

  // res block: out = idn + gconv(bnrelu(gconv(bnrelu(x)))); residual fused in store
  auto resblock = [&](const RB& rb, const float* x, int N, const int* nbr,
                      float* out, int l) {
    const float* idn = x;
    int lda = rb.cin;
    if (rb.Wi) {                                   // 1x1 identity-branch GEMM
      int n = N * rb.cin;
      k_cvt_bf16<<<dim3((unsigned)ceil_div(n, 256)), 256, 0, stream>>>(x, bB[l], n);
      gemm(bB[l], zp[l], rb.Wi, nullptr, 1, ib[l], N, rb.cin, rb.cout, rb.cout, 0, nullptr, 0);
      idn = ib[l];
      lda = rb.cout;
    }
    bnrelu(x, N, rb.cin, rb.g1, rb.b1, bA[l], st[l]);
    gemm(bA[l], zp[l], rb.W1, nbr, 27, hb[l], N, rb.cin, rb.cout, rb.cout, 0, nullptr, 0);
    bnrelu(hb[l], N, rb.cout, rb.g2, rb.b2, bA[l], st[l]);
    gemm(bA[l], zp[l], rb.W2, nbr, 27, out, N, rb.cout, rb.cout, rb.cout, 0, idn, lda);
  };

  // -------- U-block recursion --------
  std::function<float*(int)> run = [&](int l) -> float* {
    int N = NL[l], C = CL[l];
    float* cur = xb[l];
    float* alt = yb[l];
    for (int i = 0; i < 2; ++i) {
      resblock(lvl[l].blocks[i], cur, N, nbr_subm[l], alt, l);
      std::swap(cur, alt);
    }
    if (l < 3) {
      int N2 = NL[l + 1], C2 = CL[l + 1];
      float* idn = cur;
      bnrelu(cur, N, C, lvl[l].conv_g, lvl[l].conv_b, bA[l], st[l]);
      gemm(bA[l], zp[l], lvl[l].Wdown, nbr_down[l], 8, xb[l + 1], N2, C, C2, C2, 0, nullptr, 0);
      float* res = run(l + 1);
      bnrelu(res, N2, C2, lvl[l].deconv_g, lvl[l].deconv_b, bA[l + 1], st[l + 1]);
      // concat: alt[:,0:C] = idn ; alt[:,C:2C] = inv_conv(...)
      Z(alt, N * 2 * C);
      k_copy_cols<<<dim3((unsigned)ceil_div(N * C, 256)), 256, 0, stream>>>(
          idn, alt, N, C, 2 * C, 0);
      scatter(bA[l + 1], lvl[l].Wup, nbr_down[l], 8, alt, N2, C2, C, 2 * C, C);
      std::swap(cur, alt);
      for (int i = 0; i < 2; ++i) {
        resblock(lvl[l].tails[i], cur, N, nbr_subm[l], alt, l);
        std::swap(cur, alt);
      }
    }
    return cur;
  };

  // entry features -> level-0 x buffer (inputs must not be mutated)
  {
    int n = NL[0] * CL[0];
    k_copy_f32<<<dim3((unsigned)ceil_div(n, 256)), 256, 0, stream>>>(
        (const float*)d_in[0], xb[0], n);
  }
  float* result = run(0);
  {
    int n = NL[0] * CL[0];
    k_copy_f32<<<dim3((unsigned)ceil_div(n, 256)), 256, 0, stream>>>(
        result, (float*)d_out, n);
  }
}
